// RGCNNFactorization_18820546691612
// MI455X (gfx1250) — matrix operand order repaired and checked by reading, not compiled, not run on hardware
//
#include <hip/hip_runtime.h>
#include <hip/hip_bf16.h>
#include <math.h>

// ---------------------------------------------------------------------------
// Problem constants (match reference)
// ---------------------------------------------------------------------------
#define MM 10000     // rows (users) == cols (items)
#define RR 10        // factorization rank
#define QQ 32        // conv channels == lstm input
#define NHID 32      // lstm hidden
#define KCH 5        // Chebyshev order
#define TT 10        // diffusion steps
#define KR (KCH*RR)  // 50 : concatenated Chebyshev feature width
#define KRP 64       // padded Chebyshev feature width (2 WMMA K-steps)

typedef __attribute__((ext_vector_type(16))) _Float16 v16h;
typedef __attribute__((ext_vector_type(8)))  float    v8f;

// ---------------------------------------------------------------------------
// WMMA tile loaders (CDNA5 16x16x32 f16; A and B share contiguous k order k = 16*(lane/16) + s, wave32)
// All loads are UNCONDITIONAL (buffers zero-padded in workspace) so the
// compiler emits wide global_load_b128 instead of divergent masked loads.
// ---------------------------------------------------------------------------
// A (MxK tile = 16x32): lane gives M; lanes 0-15 hold K 0-15,
//                       lanes 16-31 hold K 16-31.
static __device__ inline v16h load_a16(const float* __restrict__ A, int lda,
                                       int m0, int k0) {
  int lane = threadIdx.x & 31;
  const float* p = A + (size_t)(m0 + (lane & 15)) * lda;
  int b0 = k0 + ((lane < 16) ? 0 : 16);
  int b1 = k0 + ((lane < 16) ? 8 : 24);
  v16h a;
#pragma unroll
  for (int i = 0; i < 8; ++i) a[i] = (_Float16)p[b0 + i];
#pragma unroll
  for (int i = 0; i < 8; ++i) a[8 + i] = (_Float16)p[b1 + i];
  return a;
}

// B (KxN tile = 32x16) supplied TRANSPOSED (N-major: Bt[n*ldk + k]):
// lane gives N; lanes 0-15 hold K 0-15, lanes 16-31 hold K 16-31.
// Per-lane 16 K-values are contiguous -> 4x global_load_b128.
static __device__ inline v16h load_bt16(const float* __restrict__ Bt, int ldk,
                                        int k0, int n0) {
  int lane = threadIdx.x & 31;
  const float* p =
      Bt + (size_t)(n0 + (lane & 15)) * ldk + k0 + ((lane < 16) ? 0 : 16);
  v16h b;
#pragma unroll
  for (int i = 0; i < 16; ++i) b[i] = (_Float16)p[i];
  return b;
}

static __device__ inline v8f wmma_f16(v16h a, v16h b, v8f c) {
  return __builtin_amdgcn_wmma_f32_16x16x32_f16(
      /*neg_a=*/false, a, /*neg_b=*/false, b,
      /*c_mod=*/(short)0, c, /*reuse_a=*/false, /*reuse_b=*/false);
}

static __device__ inline float sigmoidf(float x) {
  return 1.0f / (1.0f + expf(-x));
}

// ---------------------------------------------------------------------------
// Graph kernels (ChebConv sparse part)
// ---------------------------------------------------------------------------
__global__ void deg_kernel(const int* __restrict__ src, float* deg, int E) {
  int e = blockIdx.x * blockDim.x + threadIdx.x;
  if (e < E) atomicAdd(&deg[src[e]], 1.0f);
}

__global__ void dinv_kernel(float* deg, int n) {
  int i = blockIdx.x * blockDim.x + threadIdx.x;
  if (i < n) {
    float d = deg[i];
    deg[i] = (d > 0.0f) ? rsqrtf(fmaxf(d, 1.0f)) : 0.0f;
  }
}

// tx[:, 0:R] = X  (X is [n, R] row-major; tx row stride KRP)
__global__ void copy_block_kernel(const float* __restrict__ X, float* tx, int n) {
  int i = blockIdx.x * blockDim.x + threadIdx.x;
  if (i < n * RR) {
    int m = i / RR, r = i - m * RR;
    tx[(size_t)m * KRP + r] = X[i];
  }
}

// tx[:, kc:kc+R] = -tx[:, kp:kp+R]
__global__ void init_neg_kernel(float* tx, int kc, int kp, int n) {
  int i = blockIdx.x * blockDim.x + threadIdx.x;
  if (i < n * RR) {
    int m = i / RR, r = i - m * RR;
    tx[(size_t)m * KRP + kc + r] = -tx[(size_t)m * KRP + kp + r];
  }
}

// tx[dst, to:to+R] += scale * (-dinv[src]*dinv[dst]) * tx[src, from:from+R]
__global__ void prop_kernel(const int* __restrict__ src,
                            const int* __restrict__ dst,
                            const float* __restrict__ dinv, float* tx,
                            int from_col, int to_col, float scale, int E) {
  int e = blockIdx.x * blockDim.x + threadIdx.x;
  if (e >= E) return;
  int s = src[e], d = dst[e];
  float w = -dinv[s] * dinv[d] * scale;
  const float* zi = tx + (size_t)s * KRP + from_col;
  float* zo = tx + (size_t)d * KRP + to_col;
#pragma unroll
  for (int r = 0; r < RR; ++r) atomicAdd(&zo[r], w * zi[r]);
}

// hconv_w [50,32] -> zero-padded transpose [32,64]  (wpt[n*64+k] = w[k*32+n])
__global__ void pad_w_transpose_kernel(const float* __restrict__ w,
                                       float* __restrict__ wpt) {
  int i = blockIdx.x * blockDim.x + threadIdx.x;
  if (i < KR * QQ) {
    int k = i / QQ, n = i - k * QQ;
    wpt[(size_t)n * KRP + k] = w[i];
  }
}

// dh_w [10,32] -> zero-padded [16,32] (already N-major for outs @ dw.T)
__global__ void pad_dw_kernel(const float* __restrict__ dw,
                              float* __restrict__ dwp) {
  int i = blockIdx.x * blockDim.x + threadIdx.x;
  if (i < RR * QQ) dwp[i] = dw[i];
}

// ---------------------------------------------------------------------------
// WMMA GEMMs (one wave per 16x16 D tile; EXEC all-ones through the WMMA)
// ---------------------------------------------------------------------------
// Htilde = sigmoid( tx[10000,64p] @ wpt.T + bias )   grid(625, 2), block 32
__global__ void cheb_gemm_sigmoid(const float* __restrict__ tx,
                                  const float* __restrict__ wpt, // [32,64]
                                  const float* __restrict__ bias,
                                  float* __restrict__ out) {
  int m0 = blockIdx.x * 16;
  int n0 = blockIdx.y * 16;
  v8f acc = {};
  acc = wmma_f16(load_a16(tx, KRP, m0, 0),  load_bt16(wpt, KRP, 0, n0),  acc);
  acc = wmma_f16(load_a16(tx, KRP, m0, 32), load_bt16(wpt, KRP, 32, n0), acc);
  int lane = threadIdx.x & 31;
  int n = n0 + (lane & 15);
  int mb = (lane < 16) ? 0 : 8;
#pragma unroll
  for (int r = 0; r < 8; ++r) {
    int m = m0 + mb + r;
    out[(size_t)m * QQ + n] = sigmoidf(acc[r] + bias[n]);
  }
}

// xg = x[10000,32] @ wih.T + bih   (wih [128,32] is already N-major Bt)
// grid(625, 8), block 32
__global__ void xg_gemm(const float* __restrict__ x,
                        const float* __restrict__ wih,
                        const float* __restrict__ bih,
                        float* __restrict__ xg) {
  int m0 = blockIdx.x * 16;
  int n0 = blockIdx.y * 16;
  v8f acc = {};
  acc = wmma_f16(load_a16(x, QQ, m0, 0), load_bt16(wih, QQ, 0, n0), acc);
  int lane = threadIdx.x & 31;
  int n = n0 + (lane & 15);
  int mb = (lane < 16) ? 0 : 8;
#pragma unroll
  for (int r = 0; r < 8; ++r) {
    int m = m0 + mb + r;
    xg[(size_t)m * (4 * NHID) + n] = acc[r] + bih[n];
  }
}

// Out = Base + tanh( outs[10000,32] @ dwp.T + db )  (dwp [16,32] padded)
// grid(625), block 32
__global__ void proj_gemm(const float* __restrict__ outs,
                          const float* __restrict__ dwp,
                          const float* __restrict__ db,
                          const float* __restrict__ base,
                          float* __restrict__ out) {
  int m0 = blockIdx.x * 16;
  v8f acc = {};
  acc = wmma_f16(load_a16(outs, QQ, m0, 0), load_bt16(dwp, QQ, 0, 0), acc);
  int lane = threadIdx.x & 31;
  int n = lane & 15;
  int mb = (lane < 16) ? 0 : 8;
  if (n < RR) {
#pragma unroll
    for (int r = 0; r < 8; ++r) {
      int m = m0 + mb + r;
      out[(size_t)m * RR + n] =
          base[(size_t)m * RR + n] + tanhf(acc[r] + db[n]);
    }
  }
}

// ---------------------------------------------------------------------------
// Sequential LSTM: single block, 128 threads (one gate-row each).
// Runs all 2*T passes; records outs for the final H-pass and W-pass.
// ---------------------------------------------------------------------------
__global__ void lstm_seq_kernel(const float* __restrict__ xgH,
                                const float* __restrict__ xgW,
                                const float* __restrict__ whh,
                                const float* __restrict__ bhh,
                                float* __restrict__ outsH,
                                float* __restrict__ outsW) {
  __shared__ float sh[NHID];
  __shared__ float sc[NHID];
  __shared__ float sg[4 * NHID];
  const int r = threadIdx.x;  // 0..127 gate row

  float wrow[NHID];
#pragma unroll
  for (int j = 0; j < NHID; ++j) wrow[j] = whh[(size_t)r * NHID + j];
  const float br = bhh[r];

  if (r < NHID) { sh[r] = 0.0f; sc[r] = 0.0f; }
  __syncthreads();

  for (int pass = 0; pass < 2 * TT; ++pass) {
    const float* xg = (pass & 1) ? xgW : xgH;
    float* rec = (pass & 1) ? outsW : outsH;
    const bool record = (pass >= 2 * TT - 2);
    for (int t = 0; t < MM; ++t) {
      float g = xg[(size_t)t * (4 * NHID) + r] + br;
#pragma unroll
      for (int j = 0; j < NHID; ++j) g += wrow[j] * sh[j];
      sg[r] = g;
      __syncthreads();
      if (r < NHID) {
        float gi = sg[r];
        float gf = sg[NHID + r];
        float gg = sg[2 * NHID + r];
        float go = sg[3 * NHID + r];
        float c = sigmoidf(gf) * sc[r] + sigmoidf(gi) * tanhf(gg);
        float h = sigmoidf(go) * tanhf(c);
        sc[r] = c;
        sh[r] = h;
        if (record) rec[(size_t)t * NHID + r] = h;
      }
      __syncthreads();
    }
  }
}

// ---------------------------------------------------------------------------
// Host launcher
// ---------------------------------------------------------------------------
extern "C" void kernel_launch(void* const* d_in, const int* in_sizes, int n_in,
                              void* d_out, int out_size, void* d_ws, size_t ws_size,
                              hipStream_t stream) {
  const float* H  = (const float*)d_in[0];   // [10000,10]
  const float* W  = (const float*)d_in[1];   // [10000,10]
  const int* HA   = (const int*)d_in[2];     // [2, 160000]
  const int* WA   = (const int*)d_in[3];     // [2, 160000]
  const float* hconv_w = (const float*)d_in[4];   // [5,10,32] == [50,32]
  const float* hconv_b = (const float*)d_in[5];   // [32]
  const float* wconv_w = (const float*)d_in[6];
  const float* wconv_b = (const float*)d_in[7];
  const float* lstm_wih = (const float*)d_in[8];  // [128,32]
  const float* lstm_whh = (const float*)d_in[9];  // [128,32]
  const float* lstm_bih = (const float*)d_in[10]; // [128]
  const float* lstm_bhh = (const float*)d_in[11]; // [128]
  const float* dh_w = (const float*)d_in[12];     // [10,32]
  const float* dh_b = (const float*)d_in[13];     // [10]
  const float* dw_w = (const float*)d_in[14];
  const float* dw_b = (const float*)d_in[15];

  const int E = in_sizes[2] / 2;  // 160000 directed edges
  const int* HAsrc = HA;
  const int* HAdst = HA + E;
  const int* WAsrc = WA;
  const int* WAdst = WA + E;

  // workspace layout (floats); zero-padded region first
  float* f = (float*)d_ws;
  float* dinvH  = f;                       // 10000
  float* dinvW  = dinvH + MM;              // 10000
  float* txH    = dinvW + MM;              // 10000*64 (zero-padded K)
  float* txW    = txH + (size_t)MM * KRP;  // 10000*64
  float* wptH   = txW + (size_t)MM * KRP;  // 32*64
  float* wptW   = wptH + QQ * KRP;         // 32*64
  float* dwpH   = wptW + QQ * KRP;         // 16*32
  float* dwpW   = dwpH + 16 * QQ;          // 16*32
  float* zend   = dwpW + 16 * QQ;
  float* Htilde = zend;                    // 10000*32
  float* Wtilde = Htilde + (size_t)MM * QQ;
  float* xgH    = Wtilde + (size_t)MM * QQ;        // 10000*128
  float* xgW    = xgH + (size_t)MM * 4 * NHID;     // 10000*128
  float* outsH  = xgW + (size_t)MM * 4 * NHID;     // 10000*32
  float* outsW  = outsH + (size_t)MM * QQ;         // 10000*32

  // zero atomic-accumulated + padded regions
  size_t zero_floats = (size_t)(zend - f);
  hipMemsetAsync(d_ws, 0, zero_floats * sizeof(float), stream);

  const int TB = 256;
  const int egrid = (E + TB - 1) / TB;
  const int ngrid = (MM + TB - 1) / TB;
  const int nrgrid = (MM * RR + TB - 1) / TB;
  const int wgrid = (KR * QQ + TB - 1) / TB;

  // degrees -> dinv
  deg_kernel<<<egrid, TB, 0, stream>>>(HAsrc, dinvH, E);
  deg_kernel<<<egrid, TB, 0, stream>>>(WAsrc, dinvW, E);
  dinv_kernel<<<ngrid, TB, 0, stream>>>(dinvH, MM);
  dinv_kernel<<<ngrid, TB, 0, stream>>>(dinvW, MM);

  // pad/transpose small weight matrices for clean WMMA B-tile loads
  pad_w_transpose_kernel<<<wgrid, TB, 0, stream>>>(hconv_w, wptH);
  pad_w_transpose_kernel<<<wgrid, TB, 0, stream>>>(wconv_w, wptW);
  pad_dw_kernel<<<wgrid, TB, 0, stream>>>(dh_w, dwpH);
  pad_dw_kernel<<<wgrid, TB, 0, stream>>>(dw_w, dwpW);

  // Chebyshev basis blocks: tx0=X, tx1=L x, tx_k = 2 L tx_{k-1} - tx_{k-2}
  copy_block_kernel<<<nrgrid, TB, 0, stream>>>(H, txH, MM);
  copy_block_kernel<<<nrgrid, TB, 0, stream>>>(W, txW, MM);
  prop_kernel<<<egrid, TB, 0, stream>>>(HAsrc, HAdst, dinvH, txH, 0, RR, 1.0f, E);
  prop_kernel<<<egrid, TB, 0, stream>>>(WAsrc, WAdst, dinvW, txW, 0, RR, 1.0f, E);
  for (int k = 2; k < KCH; ++k) {
    init_neg_kernel<<<nrgrid, TB, 0, stream>>>(txH, k * RR, (k - 2) * RR, MM);
    prop_kernel<<<egrid, TB, 0, stream>>>(HAsrc, HAdst, dinvH, txH,
                                          (k - 1) * RR, k * RR, 2.0f, E);
    init_neg_kernel<<<nrgrid, TB, 0, stream>>>(txW, k * RR, (k - 2) * RR, MM);
    prop_kernel<<<egrid, TB, 0, stream>>>(WAsrc, WAdst, dinvW, txW,
                                          (k - 1) * RR, k * RR, 2.0f, E);
  }

  // Dense Chebyshev transform + sigmoid (WMMA)
  dim3 gcheb(MM / 16, QQ / 16);
  cheb_gemm_sigmoid<<<gcheb, 32, 0, stream>>>(txH, wptH, hconv_b, Htilde);
  cheb_gemm_sigmoid<<<gcheb, 32, 0, stream>>>(txW, wptW, wconv_b, Wtilde);

  // Loop-invariant input-gate precompute (WMMA)
  dim3 gxg(MM / 16, (4 * NHID) / 16);
  xg_gemm<<<gxg, 32, 0, stream>>>(Htilde, lstm_wih, lstm_bih, xgH);
  xg_gemm<<<gxg, 32, 0, stream>>>(Wtilde, lstm_wih, lstm_bih, xgW);

  // Sequential 2*T-pass LSTM (single block — latency-bound recurrence)
  lstm_seq_kernel<<<1, 4 * NHID, 0, stream>>>(xgH, xgW, lstm_whh, lstm_bhh,
                                              outsH, outsW);

  // Final projections (WMMA) into d_out: [Hout | Wout]
  float* Hout = (float*)d_out;
  float* Wout = Hout + (size_t)MM * RR;
  proj_gemm<<<MM / 16, 32, 0, stream>>>(outsH, dwpH, dh_b, H, Hout);
  proj_gemm<<<MM / 16, 32, 0, stream>>>(outsW, dwpW, dw_b, W, Wout);
}
